// TSI_Encoder_15633680957655
// MI455X (gfx1250) — matrix operand order repaired and checked by reading, not compiled, not run on hardware
//
#include <hip/hip_runtime.h>
#include <hip/hip_bf16.h>

typedef _Float16 h16;
typedef __attribute__((ext_vector_type(16))) _Float16 v16h;
typedef __attribute__((ext_vector_type(8)))  _Float16 v8h;
typedef __attribute__((ext_vector_type(8)))  float    v8f;
typedef __attribute__((ext_vector_type(4)))  float    v4f;

#define WAVES  8
#define MTILES 9   // 144 rows = 9 tiles of 16

// ---- LDS layout (bytes) --------------------------------------------------
// feat  f16 144 x 608 (stride 608)          [0      .. 175104)
//   h1  f16 144 x 392 (alias feat)          [0      .. 112896)
//   wm  f32  16 x 144 (alias feat, high)    [163840 .. 173056)
// h0    f16 144 x 520 (stride 520)          [175104 .. 324864)
// xv    f32 176                             [324864 .. 325568)
// outAcc f32 256                            [325568 .. 326592)
#define SMEM_BYTES 326592

// A fragment (16x32 f16) from LDS: lane holds M=lane%16,
// K = k0 + (lane/16)*16 + {0..15} contiguous.
__device__ __forceinline__ v16h load_a_lds(const h16* __restrict__ buf, int strideH,
                                           int m0, int k0, int lane) {
  const h16* p = buf + (m0 + (lane & 15)) * strideH + k0 + ((lane >> 4) << 4);
  union { v16h v; v8h h[2]; } u;
  u.h[0] = *(const v8h*)p;
  u.h[1] = *(const v8h*)(p + 8);
  return u.v;
}

// B fragment (32x16 f16) from global fp32 weights (row-major O x K):
// lane holds N-column = weight row `row`, K = k0 + (lane/16)*16 + 0..15 contiguous.
__device__ __forceinline__ v16h load_b_full(const float* __restrict__ W, int kdim,
                                            int row, int k0, int lane) {
  const float* p = W + (long)row * kdim + k0 + ((lane >> 4) << 4);
  v4f f0 = *(const v4f*)(p + 0);
  v4f f1 = *(const v4f*)(p + 4);
  v4f f2 = *(const v4f*)(p + 8);
  v4f f3 = *(const v4f*)(p + 12);
  v16h r;
#pragma unroll
  for (int j = 0; j < 4; ++j) {
    r[j]      = (h16)f0[j];
    r[4 + j]  = (h16)f1[j];
    r[8 + j]  = (h16)f2[j];
    r[12 + j] = (h16)f3[j];
  }
  return r;
}

// One MLP layer as WMMA GEMM: D[144 x N] = relu(A[144 x K] * W^T + bias).
// Each wave owns NPW N-tiles; B K-strip cached in registers; A streamed from LDS.
// K-chain split into two independent accumulator chains to pipeline the matrix unit.
template<int KT, int KDIM, int NPW, bool PARTIAL, bool LAST>
__device__ void mlp_layer(const float* __restrict__ W, const float* __restrict__ bias,
                          const h16* __restrict__ inbuf, int inStride,
                          h16* outbuf, int outStride,
                          const float* __restrict__ wm, float* outAcc,
                          int wave, int lane) {
#pragma unroll 1
  for (int ni = 0; ni < NPW; ++ni) {
    const int nt  = wave + ni * WAVES;
    const int n0  = nt * 16;
    const int row = n0 + (lane & 15);   // weight row == output channel

    v16h Bf[KT];
#pragma unroll
    for (int k = 0; k < KT; ++k) {
      if (PARTIAL && k == KT - 1) {
        const int kbase = k * 32 + ((lane >> 4) << 4);
        v16h r;
#pragma unroll
        for (int j = 0; j < 16; ++j) {
          const int kk = kbase + j;
          const float f = (kk < KDIM) ? W[(long)row * KDIM + kk] : 0.0f;
          r[j] = (h16)f;
        }
        Bf[k] = r;
      } else {
        Bf[k] = load_b_full(W, KDIM, row, k * 32, lane);
      }
    }
    const float bv = bias[row];

#pragma unroll 1
    for (int mt = 0; mt < MTILES; ++mt) {
      // two independent accumulation chains (even / odd K-tiles)
      v8f acc0 = {0.f, 0.f, 0.f, 0.f, 0.f, 0.f, 0.f, 0.f};
      v8f acc1 = {0.f, 0.f, 0.f, 0.f, 0.f, 0.f, 0.f, 0.f};
#pragma unroll
      for (int k = 0; k < KT; k += 2) {
        v16h A0 = load_a_lds(inbuf, inStride, mt * 16, k * 32, lane);
        acc0 = __builtin_amdgcn_wmma_f32_16x16x32_f16(
            false, A0, false, Bf[k], (short)0, acc0, false, false);
        if (k + 1 < KT) {
          v16h A1 = load_a_lds(inbuf, inStride, mt * 16, (k + 1) * 32, lane);
          acc1 = __builtin_amdgcn_wmma_f32_16x16x32_f16(
              false, A1, false, Bf[k + 1], (short)0, acc1, false, false);
        }
      }
      // D layout: element v of lane L is (M = mt*16 + v + 8*(L/16), N = row)
      const int rbase = mt * 16 + ((lane >> 4) << 3);
      if constexpr (LAST) {
        const int g = row >> 4;
        float csum = 0.f;
#pragma unroll
        for (int v = 0; v < 8; ++v) {
          float xx = acc0[v] + acc1[v] + bv;
          xx = xx > 0.f ? xx : 0.f;
          csum += xx * wm[g * 144 + rbase + v];
        }
        atomicAdd(&outAcc[row], csum);   // ds_add_f32
      } else {
#pragma unroll
        for (int v = 0; v < 8; ++v) {
          float xx = acc0[v] + acc1[v] + bv;
          xx = xx > 0.f ? xx : 0.f;
          outbuf[(rbase + v) * outStride + row] = (h16)xx;
        }
      }
    }
  }
}

__global__ __launch_bounds__(256)
void tsi_encoder_kernel(const float* __restrict__ x,
                        const float* __restrict__ w0, const float* __restrict__ b0,
                        const float* __restrict__ w1, const float* __restrict__ b1,
                        const float* __restrict__ w2, const float* __restrict__ b2,
                        const float* __restrict__ wmask,
                        float* __restrict__ out) {
  __shared__ __align__(16) unsigned char smem[SMEM_BYTES];
  h16*   feat   = (h16*)smem;                        // stride 608, K padded 588->608 (zeros)
  h16*   h0     = (h16*)(smem + 175104);             // stride 520
  float* xv     = (float*)(smem + 324864);
  float* outAcc = (float*)(smem + 325568);
  h16*   h1     = (h16*)smem;                        // stride 392 (aliases dead feat)
  float* wmBuf  = (float*)(smem + 163840);           // 16x144 softmax (aliases feat tail)

  const int tid  = threadIdx.x;
  const int lane = tid & 31;
  const int wave = tid >> 5;
  const int b    = blockIdx.x;

  // ---- phase 0: load x row (x[:,167]=0), zero output accumulator ---------
  if (tid < 168) xv[tid] = (tid == 167) ? 0.0f : x[(long)b * 168 + tid];
  outAcc[tid] = 0.0f;
  __syncthreads();

  // ---- phase 1: pairwise features ----------------------------------------
  // row p = h*12+w; channel = (a*14+c)*3 + {di, di/s, xj/s},
  // xi = x[a*12+h], xj = x[c*12+w], s = xi+xj+eps.
  for (int i = tid; i < 144 * 196; i += 256) {
    const int p  = i / 196;
    const int ac = i - p * 196;
    const int a  = ac / 14;
    const int c  = ac - a * 14;
    const int hh = p / 12;
    const int ww = p - hh * 12;
    const float xi = xv[a * 12 + hh];
    const float xj = xv[c * 12 + ww];
    const float s  = xi + xj + 1e-5f;
    const float rs = __builtin_amdgcn_rcpf(s);
    const float di = xj - xi;
    h16* fp = feat + p * 608 + ac * 3;
    fp[0] = (h16)di;
    fp[1] = (h16)(di * rs);
    fp[2] = (h16)(xj * rs);
  }
  for (int i = tid; i < 144 * 20; i += 256) {      // zero K padding 588..607
    const int p = i / 20;
    feat[p * 608 + 588 + (i - p * 20)] = (h16)0.f;
  }
  __syncthreads();

  // ---- phase 2: layer0  feat[144x588] -> h0[144x512] ---------------------
  mlp_layer<19, 588, 4, true,  false>(w0, b0, feat, 608, h0, 520, nullptr, nullptr, wave, lane);
  __syncthreads();

  // ---- phase 3: layer1  h0 -> h1[144x384]; softmax(weight_mask) ----------
  mlp_layer<16, 512, 3, false, false>(w1, b1, h0, 520, h1, 392, nullptr, nullptr, wave, lane);
  if (tid < 16) {
    const float* wrow = wmask + tid * 144;
    float mx = wrow[0];
    for (int p = 1; p < 144; ++p) mx = fmaxf(mx, wrow[p]);
    float sum = 0.f;
    float* dst = wmBuf + tid * 144;
    for (int p = 0; p < 144; ++p) { const float e = __expf(wrow[p] - mx); dst[p] = e; sum += e; }
    const float inv = 1.0f / sum;
    for (int p = 0; p < 144; ++p) dst[p] *= inv;
  }
  __syncthreads();

  // ---- phase 4: layer2  h1 -> weighted spatial reduction into outAcc -----
  mlp_layer<12, 384, 2, false, true >(w2, b2, h1, 392, nullptr, 0, wmBuf, outAcc, wave, lane);
  __syncthreads();

  // ---- phase 5: final relu + store ---------------------------------------
  const float r = outAcc[tid];
  out[(long)b * 256 + tid] = r > 0.f ? r : 0.f;
}

extern "C" void kernel_launch(void* const* d_in, const int* in_sizes, int n_in,
                              void* d_out, int out_size, void* d_ws, size_t ws_size,
                              hipStream_t stream) {
  (void)in_sizes; (void)n_in; (void)out_size; (void)d_ws; (void)ws_size;
  const float* x  = (const float*)d_in[0];
  const float* w0 = (const float*)d_in[1];
  const float* b0 = (const float*)d_in[2];
  const float* w1 = (const float*)d_in[3];
  const float* b1 = (const float*)d_in[4];
  const float* w2 = (const float*)d_in[5];
  const float* b2 = (const float*)d_in[6];
  const float* wm = (const float*)d_in[7];
  tsi_encoder_kernel<<<1024, 256, 0, stream>>>(x, w0, b0, w1, b1, w2, b2, wm, (float*)d_out);
}
